// RandomDelayGwAC_28123445854585
// MI455X (gfx1250) — matrix operand order repaired and checked by reading, not compiled, not run on hardware
//
#include <hip/hip_runtime.h>
#include <math.h>

// ---------------- problem constants (from reference) ----------------
#define NN   256          // nodes == runs
#define TT   1280         // steps per run
#define INF_ 128          // in_features
#define HH   256          // hidden
#define MM   128          // message size
#define OUTF 64           // out_features
#define KK   384          // H + M (GEMM reduction dim)

#define RUNS 16           // runs batched per workgroup (== WMMA N tile)
#define NWG  (NN / RUNS)  // 16 workgroups
#define WST  392          // padded LDS row stride in bf16 elems (784 B, banks spread)

// LDS partition (in ushort elems)
#define ELEM_WS   (HH * WST)              // 100352
#define ELEM_WM   (MM * WST)              // 50176
#define ELEM_ACT  (RUNS * WST)            // 6272
#define ELEM_CTRL 352                     // 48 ints + 128 uints = 704 B
#define LDS_BYTES ((ELEM_WS + ELEM_WM + ELEM_ACT + ELEM_CTRL) * 2)  // 314304

// workspace layout (bytes, 16B aligned)
#define OFF_PRED0 0                                   // N*H f32            = 256 KB
#define OFF_STATE (OFF_PRED0 + (size_t)NN*HH*4)       // N runs * N * H f32 = 64 MB
#define OFF_MSGS  (OFF_STATE + (size_t)NN*NN*HH*4)    // N * T * M bf16     = 80 MB
#define OFF_DIAG  (OFF_MSGS  + (size_t)NN*TT*MM*2)    // N*H f32            = 256 KB

typedef __attribute__((ext_vector_type(16))) __bf16        bf16x16;
typedef __attribute__((ext_vector_type(8)))  float         f32x8;
typedef __attribute__((ext_vector_type(4)))  unsigned int  u32x4;

union FragU { bf16x16 bf; u32x4 q[2]; };

__device__ __forceinline__ unsigned short f2bf(float f) {
  unsigned int u = __float_as_uint(f);
  u += 0x7fffu + ((u >> 16) & 1u);           // round-to-nearest-even
  return (unsigned short)(u >> 16);
}
__device__ __forceinline__ unsigned int pk2(float a, float b) {
  return (unsigned int)f2bf(a) | ((unsigned int)f2bf(b) << 16);
}

// -------------------- encoder: pred0 = x @ We^T + be --------------------
__global__ void rdg_encode(const float* __restrict__ x, const float* __restrict__ We,
                           const float* __restrict__ be, float* __restrict__ pred0) {
  const int n = blockIdx.x;        // node
  const int h = threadIdx.x;       // hidden unit (256 threads)
  const float* xr = x  + (size_t)n * INF_;
  const float* wr = We + (size_t)h * INF_;
  float acc = be[h];
  #pragma unroll 4
  for (int k = 0; k < INF_; k += 4) {
    float4 a = *(const float4*)(xr + k);
    float4 b = *(const float4*)(wr + k);
    acc += a.x*b.x + a.y*b.y + a.z*b.z + a.w*b.w;
  }
  pred0[(size_t)n * HH + h] = acc;
}

// -------------------- main recurrent WMMA kernel --------------------
__global__ void rdg_main(const float* __restrict__ pred0, const float* __restrict__ fmsg,
                         const float* __restrict__ Ws,    const float* __restrict__ bs,
                         const float* __restrict__ Wm,    const float* __restrict__ bm,
                         const int* __restrict__ proc,    const int* __restrict__ parent,
                         const unsigned char* __restrict__ valid,
                         float* __restrict__ state, unsigned short* __restrict__ msgs,
                         float* __restrict__ diag) {
  extern __shared__ __align__(16) unsigned short smem[];
  unsigned short* sWs  = smem;                       // [HH][WST] bf16
  unsigned short* sWm  = smem + ELEM_WS;             // [MM][WST] bf16
  unsigned short* sAct = smem + ELEM_WS + ELEM_WM;   // [RUNS][WST] bf16 (K=0..383)
  int*          sNode  = (int*)(smem + ELEM_WS + ELEM_WM + ELEM_ACT);
  int*          sPar   = sNode + RUNS;
  int*          sVal   = sPar  + RUNS;
  unsigned int* sBm    = (unsigned int*)(sVal + RUNS);   // [RUNS][8] node-updated bitmap

  const int tid  = threadIdx.x;
  const int wave = tid >> 5;
  const int lane = tid & 31;
  const int hl   = lane >> 4;        // half-wave selector
  const int ln   = lane & 15;
  const int bb   = tid >> 4;         // run index for cooperative staging
  const int jj   = tid & 15;
  const int r0   = blockIdx.x * RUNS;

  // ---- preload weights into LDS as bf16 (one row per thread) ----
  {
    const float* src = Ws + (size_t)tid * KK;
    unsigned short* dst = sWs + tid * WST;
    #pragma unroll 1
    for (int k = 0; k < KK; k += 8) {
      float4 a = *(const float4*)(src + k);
      float4 c = *(const float4*)(src + k + 4);
      u32x4 q; q.x = pk2(a.x, a.y); q.y = pk2(a.z, a.w);
               q.z = pk2(c.x, c.y); q.w = pk2(c.z, c.w);
      *(u32x4*)(dst + k) = q;
    }
    if (tid < MM) {
      const float* src2 = Wm + (size_t)tid * KK;
      unsigned short* dst2 = sWm + tid * WST;
      #pragma unroll 1
      for (int k = 0; k < KK; k += 8) {
        float4 a = *(const float4*)(src2 + k);
        float4 c = *(const float4*)(src2 + k + 4);
        u32x4 q; q.x = pk2(a.x, a.y); q.y = pk2(a.z, a.w);
                 q.z = pk2(c.x, c.y); q.w = pk2(c.z, c.w);
        *(u32x4*)(dst2 + k) = q;
      }
    }
  }
  if (tid < RUNS * 8) sBm[tid] = 0u;
  __syncthreads();

  #pragma unroll 1
  for (int t = 0; t < TT; ++t) {
    // ---- stage 1: schedule for this step ----
    if (tid < RUNS) {
      size_t off = (size_t)(r0 + tid) * TT + t;
      sNode[tid] = proc[off];
      sPar[tid]  = parent[off];
      sVal[tid]  = valid[off];
    }
    __syncthreads();

    // ---- stage 2: build activation panel [RUNS][384] bf16 = concat(pred[node], msg) ----
    {
      const int node = sNode[bb];
      const bool upd = (sBm[bb * 8 + (node >> 5)] >> (node & 31)) & 1u;
      const float* src = upd ? state + ((size_t)(r0 + bb) * NN + node) * HH
                             : pred0 + (size_t)node * HH;
      unsigned int* dst32 = (unsigned int*)(sAct + bb * WST);
      #pragma unroll
      for (int c = 0; c < 4; ++c) {
        const int k = jj * 16 + c * 4;
        float4 v = *(const float4*)(src + k);
        dst32[k / 2]     = pk2(v.x, v.y);
        dst32[k / 2 + 1] = pk2(v.z, v.w);
      }
      const int par = sPar[bb];
      if (par < 0) {
        #pragma unroll
        for (int c = 0; c < 2; ++c) {
          const int k = jj * 8 + c * 4;
          float4 v = *(const float4*)(fmsg + k);
          dst32[(HH + k) / 2]     = pk2(v.x, v.y);
          dst32[(HH + k) / 2 + 1] = pk2(v.z, v.w);
        }
      } else {
        const unsigned short* m = msgs + ((size_t)(r0 + bb) * TT + par) * MM;
        u32x4 q = *(const u32x4*)(m + jj * 8);
        *(u32x4*)(sAct + bb * WST + HH + jj * 8) = q;
      }
    }
    __syncthreads();

    // ---- stage 3: ns = relu(Ws @ act + bs)  (16 output tiles; 2 per wave) ----
    // k-loops kept rolled: full unroll made the scheduler hoist all B-fragment
    // ds_loads and spill them to scratch (global), poisoning the recurrence.
    f32x8 acc0 = {}; f32x8 acc1 = {};
    const int rowA0 = wave * 16 + ln;
    const int rowA1 = (wave + 8) * 16 + ln;
    #pragma unroll 1
    for (int kt = 0; kt < 12; ++kt) {
      const int kb = kt * 32 + hl * 16;
      FragU b;
      b.q[0] = *(const u32x4*)(sAct + ln * WST + kb);
      b.q[1] = *(const u32x4*)(sAct + ln * WST + kb + 8);
      const int ka = kt * 32 + hl * 16;
      FragU a0, a1;
      a0.q[0] = *(const u32x4*)(sWs + rowA0 * WST + ka);
      a0.q[1] = *(const u32x4*)(sWs + rowA0 * WST + ka + 8);
      a1.q[0] = *(const u32x4*)(sWs + rowA1 * WST + ka);
      a1.q[1] = *(const u32x4*)(sWs + rowA1 * WST + ka + 8);
      acc0 = __builtin_amdgcn_wmma_f32_16x16x32_bf16(false, a0.bf, false, b.bf,
                                                     (short)0, acc0, false, false);
      acc1 = __builtin_amdgcn_wmma_f32_16x16x32_bf16(false, a1.bf, false, b.bf,
                                                     (short)0, acc1, false, false);
    }
    __syncthreads();   // everyone done reading sAct K<256 before ns overwrites it

    // ---- commit ns: one ds_store_b128 + two predicated global_store_b128 per tile ----
    {
      const int col  = ln;                       // run = column of C tile
      const int node = sNode[col];
      const int vld  = sVal[col];
      float* strow = state + ((size_t)(r0 + col) * NN + node) * HH;

      const int rb0 = wave * 16 + hl * 8;
      float4 ba = *(const float4*)(bs + rb0);
      float4 bc = *(const float4*)(bs + rb0 + 4);
      float bias0[8] = {ba.x, ba.y, ba.z, ba.w, bc.x, bc.y, bc.z, bc.w};
      float v0[8];
      #pragma unroll
      for (int p = 0; p < 8; ++p) v0[p] = fmaxf(acc0[p] + bias0[p], 0.0f);
      u32x4 p0; p0.x = pk2(v0[0], v0[1]); p0.y = pk2(v0[2], v0[3]);
                p0.z = pk2(v0[4], v0[5]); p0.w = pk2(v0[6], v0[7]);
      *(u32x4*)(sAct + col * WST + rb0) = p0;
      if (vld) {
        float4 s0 = {v0[0], v0[1], v0[2], v0[3]};
        float4 s1 = {v0[4], v0[5], v0[6], v0[7]};
        *(float4*)(strow + rb0)     = s0;
        *(float4*)(strow + rb0 + 4) = s1;
      }

      const int rb1 = (wave + 8) * 16 + hl * 8;
      float4 bd_ = *(const float4*)(bs + rb1);
      float4 be_ = *(const float4*)(bs + rb1 + 4);
      float bias1[8] = {bd_.x, bd_.y, bd_.z, bd_.w, be_.x, be_.y, be_.z, be_.w};
      float v1[8];
      #pragma unroll
      for (int p = 0; p < 8; ++p) v1[p] = fmaxf(acc1[p] + bias1[p], 0.0f);
      u32x4 p1; p1.x = pk2(v1[0], v1[1]); p1.y = pk2(v1[2], v1[3]);
                p1.z = pk2(v1[4], v1[5]); p1.w = pk2(v1[6], v1[7]);
      *(u32x4*)(sAct + col * WST + rb1) = p1;
      if (vld) {
        float4 s0 = {v1[0], v1[1], v1[2], v1[3]};
        float4 s1 = {v1[4], v1[5], v1[6], v1[7]};
        *(float4*)(strow + rb1)     = s0;
        *(float4*)(strow + rb1 + 4) = s1;
      }
    }
    if (tid < RUNS && sVal[tid]) {
      const int nd = sNode[tid];
      sBm[tid * 8 + (nd >> 5)] |= (1u << (nd & 31));
    }
    __syncthreads();

    // ---- stage 4: nm = Wm @ concat(ns, msg) + bm  (8 tiles; 1 per wave) ----
    f32x8 accm = {};
    const int rowM = wave * 16 + ln;
    #pragma unroll 1
    for (int kt = 0; kt < 12; ++kt) {
      const int kb = kt * 32 + hl * 16;
      FragU b;
      b.q[0] = *(const u32x4*)(sAct + ln * WST + kb);
      b.q[1] = *(const u32x4*)(sAct + ln * WST + kb + 8);
      const int ka = kt * 32 + hl * 16;
      FragU a;
      a.q[0] = *(const u32x4*)(sWm + rowM * WST + ka);
      a.q[1] = *(const u32x4*)(sWm + rowM * WST + ka + 8);
      accm = __builtin_amdgcn_wmma_f32_16x16x32_bf16(false, a.bf, false, b.bf,
                                                     (short)0, accm, false, false);
    }
    // ---- commit nm: one predicated 16B global store per lane ----
    {
      const int col = ln;
      const int vld = sVal[col];
      const int rb  = wave * 16 + hl * 8;
      float4 ba = *(const float4*)(bm + rb);
      float4 bc = *(const float4*)(bm + rb + 4);
      float bias[8] = {ba.x, ba.y, ba.z, ba.w, bc.x, bc.y, bc.z, bc.w};
      if (vld) {
        float m0[8];
        #pragma unroll
        for (int p = 0; p < 8; ++p) m0[p] = accm[p] + bias[p];
        u32x4 pm; pm.x = pk2(m0[0], m0[1]); pm.y = pk2(m0[2], m0[3]);
                  pm.z = pk2(m0[4], m0[5]); pm.w = pk2(m0[6], m0[7]);
        unsigned short* mrow = msgs + ((size_t)(r0 + col) * TT + t) * MM;
        *(u32x4*)(mrow + rb) = pm;
      }
    }
    __threadfence();       // make state/msgs stores visible across both CUs of the WGP
    __syncthreads();
  }

  // ---- export diagonal rows: run r's state of node r ----
  {
    const int r = r0 + bb;
    const bool upd = (sBm[bb * 8 + (r >> 5)] >> (r & 31)) & 1u;
    const float* src = upd ? state + ((size_t)r * NN + r) * HH
                           : pred0 + (size_t)r * HH;
    float* dst = diag + (size_t)r * HH;
    #pragma unroll
    for (int c = 0; c < 4; ++c) {
      const int k = jj * 16 + c * 4;
      *(float4*)(dst + k) = *(const float4*)(src + k);
    }
  }
}

// -------------------- decoder: log_softmax(diag @ Wd^T + bd) --------------------
__global__ void rdg_decode(const float* __restrict__ diag, const float* __restrict__ Wd,
                           const float* __restrict__ bd, float* __restrict__ out) {
  const int i = blockIdx.x;    // run
  const int o = threadIdx.x;   // output class (64 threads)
  const float* v = diag + (size_t)i * HH;
  const float* w = Wd   + (size_t)o * HH;
  float z = bd[o];
  #pragma unroll 4
  for (int k = 0; k < HH; k += 4) {
    float4 a = *(const float4*)(v + k);
    float4 b = *(const float4*)(w + k);
    z += a.x*b.x + a.y*b.y + a.z*b.z + a.w*b.w;
  }
  __shared__ float red[OUTF];
  red[o] = z;
  __syncthreads();
  float mx = red[0];
  for (int k = 1; k < OUTF; ++k) mx = fmaxf(mx, red[k]);
  __syncthreads();
  red[o] = __expf(z - mx);
  __syncthreads();
  float s = 0.0f;
  for (int k = 0; k < OUTF; ++k) s += red[k];
  out[(size_t)i * OUTF + o] = z - mx - __logf(s);
}

// -------------------- host entry --------------------
extern "C" void kernel_launch(void* const* d_in, const int* in_sizes, int n_in,
                              void* d_out, int out_size, void* d_ws, size_t ws_size,
                              hipStream_t stream) {
  (void)in_sizes; (void)n_in; (void)out_size; (void)ws_size;
  const float* x    = (const float*)d_in[0];
  const float* fmsg = (const float*)d_in[1];
  const float* We   = (const float*)d_in[2];
  const float* be   = (const float*)d_in[3];
  const float* Ws   = (const float*)d_in[4];
  const float* bs   = (const float*)d_in[5];
  const float* Wm   = (const float*)d_in[6];
  const float* bm   = (const float*)d_in[7];
  const float* Wd   = (const float*)d_in[8];
  const float* bd   = (const float*)d_in[9];
  const int* proc   = (const int*)d_in[10];
  const int* parent = (const int*)d_in[11];
  const unsigned char* valid = (const unsigned char*)d_in[12];

  char* ws = (char*)d_ws;
  float*          pred0 = (float*)(ws + OFF_PRED0);
  float*          state = (float*)(ws + OFF_STATE);
  unsigned short* msgs  = (unsigned short*)(ws + OFF_MSGS);
  float*          diag  = (float*)(ws + OFF_DIAG);

  hipFuncSetAttribute(reinterpret_cast<const void*>(rdg_main),
                      hipFuncAttributeMaxDynamicSharedMemorySize, (int)LDS_BYTES);

  rdg_encode<<<NN, HH, 0, stream>>>(x, We, be, pred0);
  rdg_main<<<NWG, 256, LDS_BYTES, stream>>>(pred0, fmsg, Ws, bs, Wm, bm,
                                            proc, parent, valid, state, msgs, diag);
  rdg_decode<<<NN, OUTF, 0, stream>>>(diag, Wd, bd, (float*)d_out);
}
